// BahdanauAttention_42803644072348
// MI455X (gfx1250) — matrix operand order repaired and checked by reading, not compiled, not run on hardware
//
#include <hip/hip_runtime.h>
#include <hip/hip_bf16.h>
#include <math.h>

// ---------------------------------------------------------------------------
// Bahdanau attention, B=32 S=2048 H=1024.
// Dominant cost: e_proj = enc[65536,1024] @ U_w^T[1024,1024] = 137 GFLOP.
// bf16 WMMA (16x16x32) ~8x the MAC rate of f32 WMMA (16x16x4) => bf16 GEMM
// with fused bias+tanh+V-dot epilogue (e_proj never materialized).
// Memory floor: encoder read twice (GEMM + context) = 512MB ~= 22us @23.3TB/s.
// Context pass split over 512 WGs (s-chunks) + f32 global atomics so the
// 256MB streaming read can actually saturate HBM (32 WGs cannot).
// ---------------------------------------------------------------------------

#define B_ 32
#define S_ 2048
#define H_ 1024
#define MT 64            // rows of flattened [B*S] per workgroup (scores GEMM)
#define LDSS 1032        // LDS row stride (bf16 elems): 2064B => 4-bank rotate/row
#define SCH 128          // s-rows per context-partial workgroup

typedef __attribute__((ext_vector_type(16))) __bf16 v16bf;
typedef __attribute__((ext_vector_type(8)))  __bf16 v8bf;
typedef __attribute__((ext_vector_type(8)))  float  v8f;

static __device__ inline v16bf cat8(v8bf lo, v8bf hi) {
    return __builtin_shufflevector(lo, hi, 0,1,2,3,4,5,6,7,8,9,10,11,12,13,14,15);
}
static __device__ inline v8f wmma_bf16(v16bf a, v16bf b, v8f c) {
    // (neg_a, A, neg_b, B, c_mod, C, reuse_a, reuse_b)
    return __builtin_amdgcn_wmma_f32_16x16x32_bf16(false, a, false, b, (short)0, c, false, false);
}

// ---- Kernel 0: U_w fp32 -> bf16 (2MB, stays L2 resident for all WGs) ------
__global__ void cvt_bf16_kernel(const float* __restrict__ src,
                                __bf16* __restrict__ dst, int n) {
    int i = blockIdx.x * blockDim.x + threadIdx.x;
    int stride = gridDim.x * blockDim.x;
    for (; i < n; i += stride) dst[i] = (__bf16)src[i];
}

// ---- Kernel A: hb[b,k] = hidden[b,:]·W_w[k,:] + W_b[k] + U_b[k] (67 MFLOP)
__global__ void hproj_kernel(const float* __restrict__ hidden,
                             const float* __restrict__ Ww,
                             const float* __restrict__ Wb,
                             const float* __restrict__ Ub,
                             float* __restrict__ hb) {
    int gid = blockIdx.x * blockDim.x + threadIdx.x;  // 32768 threads
    int b = gid & (B_ - 1);                           // lanes span b => W row broadcast
    int k = gid >> 5;
    const float4* w4 = (const float4*)(Ww + (size_t)k * H_);
    const float4* h4 = (const float4*)(hidden + (size_t)b * H_);
    float acc = 0.f;
    #pragma unroll 4
    for (int j = 0; j < H_ / 4; ++j) {
        float4 w = w4[j], h = h4[j];
        acc += w.x * h.x + w.y * h.y + w.z * h.z + w.w * h.w;
    }
    hb[(size_t)b * H_ + k] = acc + Wb[k] + Ub[k];
}

// ---- Kernel B: fused bf16 GEMM + tanh + V-dot -> raw scores [B,S] ---------
__global__ __launch_bounds__(256, 1)
void scores_kernel(const float* __restrict__ enc,
                   const __bf16* __restrict__ Ubf,
                   const float* __restrict__ hb,
                   const float* __restrict__ Vw,
                   const float* __restrict__ Vb,
                   float* __restrict__ scores) {
    __shared__ __bf16 sA[MT * LDSS];   // 64 x 1024 bf16 tile, padded (129KB)
    __shared__ float  sScore[MT];

    const int tid = threadIdx.x;
    const int m0  = blockIdx.x * MT;   // MT | S => single batch per WG
    const int b   = m0 / S_;

    if (tid < MT) sScore[tid] = 0.f;

    // Stage encoder tile: fp32 global -> bf16 LDS (coalesced float4 loads)
    const float4* e4 = (const float4*)(enc + (size_t)m0 * H_);
    #pragma unroll 4
    for (int it = 0; it < (MT * H_) / (256 * 4); ++it) {
        int idx = (it * 256 + tid) * 4;
        int row = idx >> 10, col = idx & (H_ - 1);
        float4 v = e4[idx >> 2];
        __bf16* p = &sA[row * LDSS + col];
        p[0] = (__bf16)v.x; p[1] = (__bf16)v.y;
        p[2] = (__bf16)v.z; p[3] = (__bf16)v.w;
    }
    __syncthreads();

    const int wave    = tid >> 5;
    const int lane    = tid & 31;
    const int wr      = wave >> 2;      // 0..1 : 32-row stripe
    const int wc      = wave & 3;       // 0..3 : 256-col stripe
    const int rowBase = wr * 32;
    const int laneLo  = lane & 15;
    const int half    = lane >> 4;      // lane group: K-half selector

    float rp[2][8];
    #pragma unroll
    for (int rf = 0; rf < 2; ++rf)
        #pragma unroll
        for (int i = 0; i < 8; ++i) rp[rf][i] = 0.f;

    for (int chunk = 0; chunk < 2; ++chunk) {
        const int nBase = wc * 256 + chunk * 128;
        v8f acc[2][8];
        #pragma unroll
        for (int rf = 0; rf < 2; ++rf)
            #pragma unroll
            for (int t = 0; t < 8; ++t)
                #pragma unroll
                for (int i = 0; i < 8; ++i) acc[rf][t][i] = 0.f;

        for (int k = 0; k < H_; k += 32) {
            // A-frags (ISA 16-bit A layout: lanes0-15 K{0..7,16..23}, 16-31 +8)
            v16bf a[2];
            #pragma unroll
            for (int rf = 0; rf < 2; ++rf) {
                int row = rowBase + rf * 16 + laneLo;
                const __bf16* base = &sA[row * LDSS + k + half * 8];
                v8bf lo = *(const v8bf*)base;
                v8bf hi = *(const v8bf*)(base + 16);
                a[rf] = cat8(lo, hi);
            }
            // B-frags: col = lane%16, K{0..7,16..23} lanes0-15, +8 for 16-31
            #pragma unroll
            for (int t = 0; t < 8; ++t) {
                int n = nBase + t * 16 + laneLo;
                const __bf16* bb = Ubf + (size_t)n * H_ + k + half * 8;
                v8bf lo = *(const v8bf*)bb;
                v8bf hi = *(const v8bf*)(bb + 16);
                v16bf bf = cat8(lo, hi);
                acc[0][t] = wmma_bf16(a[0], bf, acc[0][t]);
                acc[1][t] = wmma_bf16(a[1], bf, acc[1][t]);
            }
        }

        // Fused epilogue: tanh(e_proj + h_proj + U_b) * V_w, reduce over n
        #pragma unroll
        for (int t = 0; t < 8; ++t) {
            int n = nBase + t * 16 + laneLo;
            float hbv = hb[(size_t)b * H_ + n];
            float vw  = Vw[n];
            #pragma unroll
            for (int rf = 0; rf < 2; ++rf)
                #pragma unroll
                for (int i = 0; i < 8; ++i)
                    rp[rf][i] += tanhf(acc[rf][t][i] + hbv) * vw;
        }
    }

    // Reduce across 16-lane column group (wave32 shuffles)
    #pragma unroll
    for (int m = 1; m <= 8; m <<= 1)
        #pragma unroll
        for (int rf = 0; rf < 2; ++rf)
            #pragma unroll
            for (int i = 0; i < 8; ++i)
                rp[rf][i] += __shfl_xor(rp[rf][i], m, 32);

    if (laneLo == 0) {
        #pragma unroll
        for (int rf = 0; rf < 2; ++rf)
            #pragma unroll
            for (int i = 0; i < 8; ++i)
                atomicAdd(&sScore[rowBase + rf * 16 + half * 8 + i], rp[rf][i]);
    }
    __syncthreads();
    if (tid < MT) scores[m0 + tid] = sScore[tid] + Vb[0];
}

// ---- Kernel C1: softmax over S (scores are tiny: 256KB) + zero context ----
__global__ __launch_bounds__(256)
void softmax_kernel(const float* __restrict__ scores,
                    float* __restrict__ attn_out,
                    float* __restrict__ ctx_out) {
    __shared__ float sAttn[S_];
    __shared__ float red[256];
    const int b = blockIdx.x, tid = threadIdx.x;
    const float* sc = scores + (size_t)b * S_;

    float lmax = -INFINITY;
    for (int i = tid; i < S_; i += 256) { float v = sc[i]; sAttn[i] = v; lmax = fmaxf(lmax, v); }
    red[tid] = lmax; __syncthreads();
    for (int off = 128; off; off >>= 1) {
        if (tid < off) red[tid] = fmaxf(red[tid], red[tid + off]);
        __syncthreads();
    }
    float gmax = red[0]; __syncthreads();

    float lsum = 0.f;
    for (int i = tid; i < S_; i += 256) { float e = __expf(sAttn[i] - gmax); sAttn[i] = e; lsum += e; }
    red[tid] = lsum; __syncthreads();
    for (int off = 128; off; off >>= 1) {
        if (tid < off) red[tid] += red[tid + off];
        __syncthreads();
    }
    float inv = 1.f / red[0];

    for (int i = tid; i < S_; i += 256)
        attn_out[(size_t)b * S_ + i] = sAttn[i] * inv;

    // zero-init context accumulator for kernel C2's atomics
    ((float4*)(ctx_out + (size_t)b * H_))[tid] = make_float4(0.f, 0.f, 0.f, 0.f);
}

// ---- Kernel C2: context partials. 512 WGs stream 256MB => HBM-saturating --
// block (b, c): rows s in [c*SCH, (c+1)*SCH); thread t owns h = 4t..4t+3.
__global__ __launch_bounds__(256)
void context_kernel(const float* __restrict__ enc,
                    const float* __restrict__ attn,
                    float* __restrict__ ctx_out) {
    __shared__ float sAttn[SCH];
    const int tid = threadIdx.x;
    const int b   = blockIdx.x / (S_ / SCH);
    const int s0  = (blockIdx.x % (S_ / SCH)) * SCH;

    if (tid < SCH) sAttn[tid] = attn[(size_t)b * S_ + s0 + tid];
    __syncthreads();

    const float4* e4 = (const float4*)(enc + ((size_t)b * S_ + s0) * H_);
    float cx = 0.f, cy = 0.f, cz = 0.f, cw = 0.f;
    #pragma unroll 4
    for (int s = 0; s < SCH; ++s) {
        float a = sAttn[s];
        float4 v = e4[(size_t)s * (H_ / 4) + tid];   // coalesced 128B/8 lanes
        cx += a * v.x; cy += a * v.y; cz += a * v.z; cw += a * v.w;
    }
    float* c = ctx_out + (size_t)b * H_ + tid * 4;
    atomicAdd(c + 0, cx);      // global_atomic_add_f32, no-return
    atomicAdd(c + 1, cy);
    atomicAdd(c + 2, cz);
    atomicAdd(c + 3, cw);
}

// ---------------------------------------------------------------------------
extern "C" void kernel_launch(void* const* d_in, const int* in_sizes, int n_in,
                              void* d_out, int out_size, void* d_ws, size_t ws_size,
                              hipStream_t stream) {
    const float* hidden = (const float*)d_in[0];
    const float* enc    = (const float*)d_in[1];
    const float* Ww     = (const float*)d_in[2];
    const float* Wb     = (const float*)d_in[3];
    const float* Uw     = (const float*)d_in[4];
    const float* Ub     = (const float*)d_in[5];
    const float* Vw     = (const float*)d_in[6];
    const float* Vb     = (const float*)d_in[7];

    float* out  = (float*)d_out;
    float* ctx  = out;              // [B,1,H] = 32768 floats
    float* attn = out + B_ * H_;    // [B,1,S] = 65536 floats

    char*   ws     = (char*)d_ws;
    __bf16* Ubf    = (__bf16*)ws;                                // 2 MB
    float*  hb     = (float*)(ws + (size_t)H_ * H_ * 2);         // 128 KB
    float*  scores = hb + B_ * H_;                               // 256 KB

    cvt_bf16_kernel<<<1024, 256, 0, stream>>>(Uw, Ubf, H_ * H_);
    hproj_kernel<<<(B_ * H_) / 256, 256, 0, stream>>>(hidden, Ww, Wb, Ub, hb);
    scores_kernel<<<(B_ * S_) / MT, 256, 0, stream>>>(enc, Ubf, hb, Vw, Vb, scores);
    softmax_kernel<<<B_, 256, 0, stream>>>(scores, attn, ctx);
    context_kernel<<<B_ * (S_ / SCH), 256, 0, stream>>>(enc, attn, ctx);
}
